// DeformableTransformerDecoderLayer_36627481101040
// MI455X (gfx1250) — matrix operand order repaired and checked by reading, not compiled, not run on hardware
//
#include <hip/hip_runtime.h>
#include <math.h>

typedef _Float16 half_t;
typedef __attribute__((ext_vector_type(16))) _Float16 v16h;
typedef __attribute__((ext_vector_type(8)))  float    v8f;

#define FLAG_RELU    1
#define FLAG_OUT_F16 2
#define FLAG_A_F32   4

#define QLEN 300
#define QPAD 304
#define KPAD 320
#define CH   256
#define LEN_TOT 13294
#define ROWS 2400          // B*Q
#define MEMROWS 106352     // B*LEN

// 16-bit A-matrix (16x32) layout: lane L holds row M=L&15; half index i maps to
// K = 16*(i>>3) + 8*(L>>4) + (i&7)   (per CDNA5 ISA 7.12.2 table)
__device__ __forceinline__ int koffA(int i, int hi) {
  return ((i >> 3) << 4) + (hi << 3) + (i & 7);
}

__device__ __forceinline__ v8f wmma_f16(v16h a, v16h b, v8f c) {
  return __builtin_amdgcn_wmma_f32_16x16x32_f16(false, a, false, b, (short)0, c, false, false);
}

// ---------------------------------------------------------------- elementwise
__global__ void cast_f16_kernel(const float* __restrict__ src, half_t* __restrict__ dst, int n) {
  int i = blockIdx.x * 256 + threadIdx.x;
  if (i < n) dst[i] = (half_t)src[i];
}

__global__ void add_kernel(const float* __restrict__ a, const float* __restrict__ b,
                           float* __restrict__ o, int n) {
  int i = blockIdx.x * 256 + threadIdx.x;
  if (i < n) o[i] = a[i] + b[i];
}

// ---------------------------------------------------------------- generic GEMM
// out[M,N] = act(A[M,K] @ Bw[K,N] + bias), A f32 or f16, out f32 or f16.
// 128x64 block tile, 8 waves; each wave computes a 2x2 grid of 16x16 WMMA tiles.
// A staged through VALU (f32->f16 convert); B staged by async copy to LDS.
__global__ __launch_bounds__(256) void gemm_wmma_kernel(
    const void* __restrict__ A_, const half_t* __restrict__ Bw,
    const float* __restrict__ bias, void* __restrict__ out_,
    int M, int N, int K, int flags)
{
  __shared__ half_t As[128][34];   // [m][k] (+pad)
  __shared__ half_t Bs[32][64];    // [k][n] contiguous (async-copy target)
  const int tid = threadIdx.x, lane = tid & 31, wv = tid >> 5;
  const int hi = lane >> 4, lm = lane & 15;
  const int bm = blockIdx.y * 128, bn = blockIdx.x * 64;
  const int wm = (wv & 3) * 2;     // m-tile base (2 tiles)
  const int wn16 = (wv >> 2) * 32; // n offset within block (2 tiles = 32 cols)
  const bool a32 = (flags & FLAG_A_F32) != 0;
  const float* Af = (const float*)A_;
  const half_t* Ah = (const half_t*)A_;
  v8f acc00 = {}, acc01 = {}, acc10 = {}, acc11 = {};

  const int ar = tid >> 1, ac = (tid & 1) * 16;  // A stage: row 0..127, 16 cols
  const int br = tid >> 3, bc = (tid & 7) * 8;   // B stage: k-row 0..31, 8 n-cols (16B)
  const int am = bm + ar;
  const unsigned int bs_lds = (unsigned int)(uintptr_t)&Bs[br][bc];

  for (int kk = 0; kk < K; kk += 32) {
    // ---- B tile: async copy global -> LDS (no transform needed, f16 weights)
    {
      unsigned long long gaddr =
          (unsigned long long)(uintptr_t)(Bw + (size_t)(kk + br) * N + bn + bc);
      asm volatile("global_load_async_to_lds_b128 %0, %1, off"
                   :: "v"(bs_lds), "v"(gaddr) : "memory");
    }
    // ---- A tile: load + convert through VALU
    if (am < M) {
      if (a32) {
        const float* src = Af + (size_t)am * K + kk + ac;
        if (kk + 32 < K) __builtin_prefetch(src + 32, 0, 0);
#pragma unroll
        for (int j = 0; j < 16; ++j) As[ar][ac + j] = (half_t)src[j];
      } else {
        const half_t* src = Ah + (size_t)am * K + kk + ac;
#pragma unroll
        for (int j = 0; j < 16; ++j) As[ar][ac + j] = src[j];
      }
    } else {
#pragma unroll
      for (int j = 0; j < 16; ++j) As[ar][ac + j] = (half_t)0.f;
    }
    asm volatile("s_wait_asynccnt 0x0" ::: "memory");
    __syncthreads();

    v16h a0, a1, b0, b1;
#pragma unroll
    for (int i = 0; i < 16; ++i) {
      int ka = koffA(i, hi);
      a0[i] = As[(wm + 0) * 16 + lm][ka];
      a1[i] = As[(wm + 1) * 16 + lm][ka];
      int kb = koffA(i, hi);
      b0[i] = Bs[kb][wn16 + lm];
      b1[i] = Bs[kb][wn16 + 16 + lm];
    }
    acc00 = wmma_f16(a0, b0, acc00);
    acc01 = wmma_f16(a0, b1, acc01);
    acc10 = wmma_f16(a1, b0, acc10);
    acc11 = wmma_f16(a1, b1, acc11);
    __syncthreads();
  }

#pragma unroll
  for (int mt = 0; mt < 2; ++mt) {
#pragma unroll
    for (int r = 0; r < 8; ++r) {
      int m = bm + (wm + mt) * 16 + r + hi * 8;
      if (m >= M) continue;
#pragma unroll
      for (int nt = 0; nt < 2; ++nt) {
        const v8f& acc = mt ? (nt ? acc11 : acc10) : (nt ? acc01 : acc00);
        int n = bn + wn16 + nt * 16 + lm;
        float v = acc[r] + (bias ? bias[n] : 0.f);
        if (flags & FLAG_RELU) v = v > 0.f ? v : 0.f;
        if (flags & FLAG_OUT_F16) ((half_t*)out_)[(size_t)m * N + n] = (half_t)v;
        else                      ((float*)out_)[(size_t)m * N + n] = v;
      }
    }
  }
}

// ---------------------------------------------------------------- attention
__global__ __launch_bounds__(32) void attn_scores_kernel(
    const half_t* __restrict__ q16, const half_t* __restrict__ k16, float* __restrict__ S)
{
  const int qt = blockIdx.x, kt = blockIdx.y, bh = blockIdx.z;
  const int b = bh >> 3, h = bh & 7;
  const int lane = threadIdx.x, hi = lane >> 4, lm = lane & 15;
  const int qrow = qt * 16 + lm;
  const int krow = kt * 16 + lm;
  v16h a, bb;
#pragma unroll
  for (int i = 0; i < 16; ++i) {
    a[i]  = (qrow < QLEN) ? q16[(size_t)(b * QLEN + qrow) * CH + h * 32 + koffA(i, hi)] : (half_t)0.f;
    bb[i] = (krow < QLEN) ? k16[(size_t)(b * QLEN + krow) * CH + h * 32 + koffA(i, hi)] : (half_t)0.f;
  }
  v8f c = {};
  c = wmma_f16(a, bb, c);
  const float scale = 0.17677669529663687f;  // 1/sqrt(32)
#pragma unroll
  for (int r = 0; r < 8; ++r) {
    int row = qt * 16 + r + hi * 8;
    int col = kt * 16 + lm;
    float v = (col < QLEN) ? c[r] * scale : -1e30f;
    S[((size_t)bh * QPAD + row) * KPAD + col] = v;
  }
}

__global__ __launch_bounds__(32) void softmax_kernel(const float* __restrict__ S,
                                                     half_t* __restrict__ P)
{
  const int row = blockIdx.x, bh = blockIdx.y;
  const int lane = threadIdx.x;
  const float* s = S + ((size_t)bh * QPAD + row) * KPAD;
  half_t* p = P + ((size_t)bh * QPAD + row) * KPAD;
  float vals[10];
  float m = -1e30f;
#pragma unroll
  for (int k = 0; k < 10; ++k) {
    int col = lane + 32 * k;
    float v = (col < QLEN) ? s[col] : -1e30f;
    vals[k] = v;
    m = fmaxf(m, v);
  }
#pragma unroll
  for (int o = 16; o > 0; o >>= 1) m = fmaxf(m, __shfl_xor(m, o));
  float sum = 0.f;
#pragma unroll
  for (int k = 0; k < 10; ++k) {
    int col = lane + 32 * k;
    float e = (col < QLEN) ? __expf(vals[k] - m) : 0.f;
    vals[k] = e;
    sum += e;
  }
#pragma unroll
  for (int o = 16; o > 0; o >>= 1) sum += __shfl_xor(sum, o);
  float inv = 1.f / sum;
#pragma unroll
  for (int k = 0; k < 10; ++k) p[lane + 32 * k] = (half_t)(vals[k] * inv);
}

__global__ __launch_bounds__(32) void attn_out_kernel(
    const half_t* __restrict__ P, const half_t* __restrict__ vv, half_t* __restrict__ ao16)
{
  const int dt = blockIdx.x, qt = blockIdx.y, bh = blockIdx.z;
  const int b = bh >> 3, h = bh & 7;
  const int lane = threadIdx.x, hi = lane >> 4, lm = lane & 15;
  const half_t* prow = P + ((size_t)bh * QPAD + qt * 16 + lm) * KPAD;
  v8f c = {};
  for (int kk = 0; kk < KPAD; kk += 32) {
    v16h a, bb;
#pragma unroll
    for (int i = 0; i < 16; ++i) {
      a[i] = prow[kk + koffA(i, hi)];
      int key = kk + koffA(i, hi);
      bb[i] = (key < QLEN) ? vv[(size_t)(b * QLEN + key) * CH + h * 32 + dt * 16 + lm] : (half_t)0.f;
    }
    c = wmma_f16(a, bb, c);
  }
#pragma unroll
  for (int r = 0; r < 8; ++r) {
    int q = qt * 16 + r + hi * 8;
    if (q < QLEN)
      ao16[(size_t)(b * QLEN + q) * CH + h * 32 + dt * 16 + lm] = (half_t)c[r];
  }
}

// ---------------------------------------------------------------- layernorm
__global__ __launch_bounds__(256) void ln_res_kernel(
    const float* __restrict__ x, const float* __restrict__ y,
    const float* __restrict__ g, const float* __restrict__ be,
    const float* __restrict__ qpe, float* __restrict__ t_out,
    float* __restrict__ query_out)
{
  __shared__ float red[8];
  const int row = blockIdx.x;
  const int c = threadIdx.x, lane = c & 31, wv = c >> 5;
  size_t idx = (size_t)row * CH + c;
  float v = x[idx] + y[idx];
  float s = v;
#pragma unroll
  for (int o = 16; o > 0; o >>= 1) s += __shfl_xor(s, o);
  if (lane == 0) red[wv] = s;
  __syncthreads();
  float mean = 0.f;
#pragma unroll
  for (int w = 0; w < 8; ++w) mean += red[w];
  mean *= (1.f / 256.f);
  float d = v - mean;
  float s2 = d * d;
#pragma unroll
  for (int o = 16; o > 0; o >>= 1) s2 += __shfl_xor(s2, o);
  __syncthreads();
  if (lane == 0) red[wv] = s2;
  __syncthreads();
  float var = 0.f;
#pragma unroll
  for (int w = 0; w < 8; ++w) var += red[w];
  var *= (1.f / 256.f);
  float t = d * rsqrtf(var + 1e-5f) * g[c] + be[c];
  t_out[idx] = t;
  if (query_out) query_out[idx] = t + qpe[idx];
}

// ---------------------------------------------------------------- deformable
__global__ __launch_bounds__(128) void sample_prep_kernel(
    const float* __restrict__ attnw, const float* __restrict__ offs,
    const float* __restrict__ refp, float* __restrict__ locs, float* __restrict__ aw)
{
  const int row = blockIdx.x;               // b*Q + q
  const int t = threadIdx.x;
  const int h = t >> 4, j = t & 15, l = j >> 2;
  const float lwh[4] = {100.f, 50.f, 25.f, 13.f};
  float logit = attnw[(size_t)row * 128 + h * 16 + j];
  float m = logit;
#pragma unroll
  for (int o = 8; o > 0; o >>= 1) m = fmaxf(m, __shfl_xor(m, o, 16));
  float e = __expf(logit - m);
  float sum = e;
#pragma unroll
  for (int o = 8; o > 0; o >>= 1) sum += __shfl_xor(sum, o, 16);
  float w = e / sum;
  float ox = offs[(size_t)row * CH + h * 32 + j * 2 + 0];
  float oy = offs[(size_t)row * CH + h * 32 + j * 2 + 1];
  float rx = refp[(size_t)row * 8 + l * 2 + 0];
  float ry = refp[(size_t)row * 8 + l * 2 + 1];
  size_t o = ((size_t)row * 8 + h) * 16 + j;
  locs[o * 2 + 0] = rx + ox / lwh[l];
  locs[o * 2 + 1] = ry + oy / lwh[l];
  aw[o] = w;
}

__global__ __launch_bounds__(256) void deform_sample_kernel(
    const half_t* __restrict__ val16, const float* __restrict__ locs,
    const float* __restrict__ aw, half_t* __restrict__ samp16)
{
  const int row = blockIdx.x;               // b*Q + q
  const int b = row / QLEN;
  const int t = threadIdx.x;
  const int h = t >> 5, d = t & 31;
  const int LW[4] = {100, 50, 25, 13};
  const int LS[4] = {0, 10000, 12500, 13125};
  float acc = 0.f;
  const float* lrow = locs + ((size_t)row * 8 + h) * 32;
  const float* arow = aw + ((size_t)row * 8 + h) * 16;
  const half_t* vb = val16 + (size_t)b * LEN_TOT * CH + h * 32 + d;
#pragma unroll
  for (int j = 0; j < 16; ++j) {
    int l = j >> 2;
    int W = LW[l], H = LW[l], st = LS[l];
    float x = lrow[j * 2 + 0] * (float)W - 0.5f;
    float y = lrow[j * 2 + 1] * (float)H - 0.5f;
    float x0 = floorf(x), y0 = floorf(y);
    float dx = x - x0, dy = y - y0;
    float sv = 0.f;
#pragma unroll
    for (int tap = 0; tap < 4; ++tap) {
      int xi = (int)x0 + (tap & 1);
      int yi = (int)y0 + (tap >> 1);
      float wx = (tap & 1) ? dx : 1.f - dx;
      float wy = (tap >> 1) ? dy : 1.f - dy;
      bool valid = (xi >= 0) && (xi < W) && (yi >= 0) && (yi < H);
      int cx = xi < 0 ? 0 : (xi > W - 1 ? W - 1 : xi);
      int cy = yi < 0 ? 0 : (yi > H - 1 ? H - 1 : yi);
      float gv = valid ? (float)vb[(size_t)(st + cy * W + cx) * CH] : 0.f;
      sv += wx * wy * gv;
    }
    acc += arow[j] * sv;
  }
  samp16[(size_t)row * CH + h * 32 + d] = (half_t)acc;
}

// ---------------------------------------------------------------- driver
extern "C" void kernel_launch(void* const* d_in, const int* in_sizes, int n_in,
                              void* d_out, int out_size, void* d_ws, size_t ws_size,
                              hipStream_t stream) {
  (void)in_sizes; (void)n_in; (void)out_size; (void)ws_size;
  const float* tgt   = (const float*)d_in[0];
  const float* refp  = (const float*)d_in[1];
  const float* mem   = (const float*)d_in[2];
  const float* qpe   = (const float*)d_in[3];
  const float* Wq = (const float*)d_in[4];   const float* bq = (const float*)d_in[5];
  const float* Wk = (const float*)d_in[6];   const float* bk = (const float*)d_in[7];
  const float* Wv = (const float*)d_in[8];   const float* bv = (const float*)d_in[9];
  const float* Wo = (const float*)d_in[10];  const float* bo = (const float*)d_in[11];
  const float* g1 = (const float*)d_in[12];  const float* be1 = (const float*)d_in[13];
  const float* W_off  = (const float*)d_in[14]; const float* b_off  = (const float*)d_in[15];
  const float* W_attn = (const float*)d_in[16]; const float* b_attn = (const float*)d_in[17];
  const float* W_val  = (const float*)d_in[18]; const float* b_val  = (const float*)d_in[19];
  const float* W_outp = (const float*)d_in[20]; const float* b_outp = (const float*)d_in[21];
  const float* g2 = (const float*)d_in[22];  const float* be2 = (const float*)d_in[23];
  const float* W1 = (const float*)d_in[24];  const float* b1 = (const float*)d_in[25];
  const float* W2 = (const float*)d_in[26];  const float* b2 = (const float*)d_in[27];
  const float* g3 = (const float*)d_in[28];  const float* be3 = (const float*)d_in[29];

  char* ws = (char*)d_ws;
  size_t off = 0;
  auto alloc = [&](size_t bytes) -> void* {
    void* p = ws + off;
    off = (off + bytes + 255) & ~(size_t)255;
    return p;
  };

  // f16 weights
  half_t* Wq16 = (half_t*)alloc(65536 * 2);
  half_t* Wk16 = (half_t*)alloc(65536 * 2);
  half_t* Wv16 = (half_t*)alloc(65536 * 2);
  half_t* Wo16 = (half_t*)alloc(65536 * 2);
  half_t* Wval16 = (half_t*)alloc(65536 * 2);
  half_t* Wout16 = (half_t*)alloc(65536 * 2);
  half_t* Woff16 = (half_t*)alloc(65536 * 2);
  half_t* Wattn16 = (half_t*)alloc(32768 * 2);
  half_t* W116 = (half_t*)alloc(262144 * 2);
  half_t* W216 = (half_t*)alloc(262144 * 2);
  // activations
  float*  qk    = (float*)alloc((size_t)ROWS * CH * 4);
  half_t* q16   = (half_t*)alloc((size_t)ROWS * CH * 2);
  half_t* k16   = (half_t*)alloc((size_t)ROWS * CH * 2);
  half_t* v16   = (half_t*)alloc((size_t)ROWS * CH * 2);
  float*  S     = (float*)alloc((size_t)64 * QPAD * KPAD * 4);
  half_t* P     = (half_t*)alloc((size_t)64 * QPAD * KPAD * 2);
  half_t* ao16  = (half_t*)alloc((size_t)ROWS * CH * 2);
  float*  sa    = (float*)alloc((size_t)ROWS * CH * 4);
  float*  t1    = (float*)alloc((size_t)ROWS * CH * 4);
  float*  query = (float*)alloc((size_t)ROWS * CH * 4);
  float*  offb  = (float*)alloc((size_t)ROWS * CH * 4);
  float*  attnw = (float*)alloc((size_t)ROWS * 128 * 4);
  float*  locs  = (float*)alloc((size_t)ROWS * 8 * 16 * 2 * 4);
  float*  awbuf = (float*)alloc((size_t)ROWS * 8 * 16 * 4);
  half_t* val16 = (half_t*)alloc((size_t)MEMROWS * CH * 2);
  half_t* samp16 = (half_t*)alloc((size_t)ROWS * CH * 2);
  float*  ca    = (float*)alloc((size_t)ROWS * CH * 4);
  float*  t2    = (float*)alloc((size_t)ROWS * CH * 4);
  half_t* h16   = (half_t*)alloc((size_t)ROWS * 1024 * 2);
  float*  ff    = (float*)alloc((size_t)ROWS * CH * 4);

  auto cast = [&](const float* s, half_t* d, int n) {
    cast_f16_kernel<<<(n + 255) / 256, 256, 0, stream>>>(s, d, n);
  };
  auto gemm = [&](const void* A, const half_t* B, const float* bias, void* out,
                  int M, int N, int K, int flags) {
    dim3 g(N / 64, (M + 127) / 128);
    gemm_wmma_kernel<<<g, 256, 0, stream>>>(A, B, bias, out, M, N, K, flags);
  };

  cast(Wq, Wq16, 65536);     cast(Wk, Wk16, 65536);
  cast(Wv, Wv16, 65536);     cast(Wo, Wo16, 65536);
  cast(W_val, Wval16, 65536); cast(W_outp, Wout16, 65536);
  cast(W_off, Woff16, 65536); cast(W_attn, Wattn16, 32768);
  cast(W1, W116, 262144);    cast(W2, W216, 262144);

  // qk = tgt + query_pos
  add_kernel<<<(ROWS * CH + 255) / 256, 256, 0, stream>>>(tgt, qpe, qk, ROWS * CH);

  // projections (f16 outputs for WMMA attention)
  gemm(qk,  Wq16, bq, q16, ROWS, CH, CH, FLAG_A_F32 | FLAG_OUT_F16);
  gemm(qk,  Wk16, bk, k16, ROWS, CH, CH, FLAG_A_F32 | FLAG_OUT_F16);
  gemm(tgt, Wv16, bv, v16, ROWS, CH, CH, FLAG_A_F32 | FLAG_OUT_F16);
  // value = memory @ W_val  (biggest GEMM; stream f32 in, f16 out)
  gemm(mem, Wval16, b_val, val16, MEMROWS, CH, CH, FLAG_A_F32 | FLAG_OUT_F16);

  // self attention
  attn_scores_kernel<<<dim3(19, 19, 64), 32, 0, stream>>>(q16, k16, S);
  softmax_kernel<<<dim3(QPAD, 64), 32, 0, stream>>>(S, P);
  attn_out_kernel<<<dim3(2, 19, 64), 32, 0, stream>>>(P, v16, ao16);
  gemm(ao16, Wo16, bo, sa, ROWS, CH, CH, 0);
  ln_res_kernel<<<ROWS, 256, 0, stream>>>(tgt, sa, g1, be1, qpe, t1, query);

  // deformable attention
  gemm(query, Woff16,  b_off,  offb,  ROWS, CH,  CH, FLAG_A_F32);
  gemm(query, Wattn16, b_attn, attnw, ROWS, 128, CH, FLAG_A_F32);
  sample_prep_kernel<<<ROWS, 128, 0, stream>>>(attnw, offb, refp, locs, awbuf);
  deform_sample_kernel<<<ROWS, 256, 0, stream>>>(val16, locs, awbuf, samp16);
  gemm(samp16, Wout16, b_outp, ca, ROWS, CH, CH, 0);
  ln_res_kernel<<<ROWS, 256, 0, stream>>>(t1, ca, g2, be2, nullptr, t2, nullptr);

  // FFN
  gemm(t2,  W116, b1, h16, ROWS, 1024, CH,   FLAG_A_F32 | FLAG_OUT_F16 | FLAG_RELU);
  gemm(h16, W216, b2, ff,  ROWS, CH,   1024, 0);
  ln_res_kernel<<<ROWS, 256, 0, stream>>>(t2, ff, g3, be3, nullptr, (float*)d_out, nullptr);
}
